// MultiheadAttention_8504035246331
// MI455X (gfx1250) — matrix operand order repaired and checked by reading, not compiled, not run on hardware
//
#include <hip/hip_runtime.h>
#include <hip/hip_bf16.h>
#include <stdint.h>

typedef __attribute__((ext_vector_type(16))) __bf16 v16bf;
typedef __attribute__((ext_vector_type(8)))  float  v8f;
typedef __attribute__((ext_vector_type(4))) unsigned int u32x4;
typedef __attribute__((ext_vector_type(8))) int          i32x8;
typedef __attribute__((ext_vector_type(4))) int          i32x4;

#define B_  4
#define S_  2048
#define D_  1024
#define H_  16
#define DH_ 64

__device__ __forceinline__ uint16_t f2bf(float f) {
    union { __bf16 b; uint16_t u; } x; x.b = (__bf16)f;   // hw v_cvt, RNE
    return x.u;
}

union AFrag { v16bf v; uint32_t u[8]; };
union BFrag { v16bf v; uint32_t u[8]; };

__device__ __forceinline__ int a_frag_k(int i, int half) {
    // 16-bit A 16x32: lanes 0-15 hold K=0..7,16..23; lanes 16-31 hold K=8..15,24..31
    return (i < 4 ? 2 * i : 16 + 2 * (i - 4)) + 8 * half;
}
__device__ __forceinline__ int b_frag_k(int i, int half) {
    // 16-bit B 32x16: lanes 0-15 hold K=0..7,16..23; lanes 16-31 hold K=8..15,24..31
    return (i < 4 ? 2 * i : 16 + 2 * (i - 4)) + 8 * half;
}

// xor-reduce over the 16-lane N group via ds_swizzle (SWAPX1/2/4/8)
__device__ __forceinline__ float red_max16(float x) {
    float t;
    t = __int_as_float(__builtin_amdgcn_ds_swizzle(__float_as_int(x), 0x041F)); x = fmaxf(x, t);
    t = __int_as_float(__builtin_amdgcn_ds_swizzle(__float_as_int(x), 0x081F)); x = fmaxf(x, t);
    t = __int_as_float(__builtin_amdgcn_ds_swizzle(__float_as_int(x), 0x101F)); x = fmaxf(x, t);
    t = __int_as_float(__builtin_amdgcn_ds_swizzle(__float_as_int(x), 0x201F)); x = fmaxf(x, t);
    return x;
}
__device__ __forceinline__ float red_sum16(float x) {
    x += __int_as_float(__builtin_amdgcn_ds_swizzle(__float_as_int(x), 0x041F));
    x += __int_as_float(__builtin_amdgcn_ds_swizzle(__float_as_int(x), 0x081F));
    x += __int_as_float(__builtin_amdgcn_ds_swizzle(__float_as_int(x), 0x101F));
    x += __int_as_float(__builtin_amdgcn_ds_swizzle(__float_as_int(x), 0x201F));
    return x;
}

// ---------------------------------------------------------------- fp32 -> bf16
__global__ void cvt_bf16_kernel(const float* __restrict__ in,
                                uint16_t* __restrict__ out, int n) {
    int i = blockIdx.x * blockDim.x + threadIdx.x;
    int stride = gridDim.x * blockDim.x;
    for (; i < n; i += stride) out[i] = f2bf(in[i]);
}

// --------------------------------- C[M,N] = (A[M,K] @ W[N,K]^T + bias) * scale
__device__ __forceinline__ void store_out(float* p, float v)    { *p = v; }
__device__ __forceinline__ void store_out(uint16_t* p, float v) { *p = f2bf(v); }

template <typename OutT>
__global__ void __launch_bounds__(256)
gemm_xWT_bias(const uint16_t* __restrict__ A, const uint16_t* __restrict__ W,
              const float* __restrict__ bias, OutT* __restrict__ C,
              int M, int N, int K, float scale) {
    __shared__ __align__(16) uint16_t As[64][32];
    __shared__ __align__(16) uint16_t Ws[64][32];

    const int tid   = threadIdx.x;
    const int lane  = tid & 31;
    const int wave  = tid >> 5;
    const int half  = lane >> 4;
    const int lan16 = lane & 15;

    const int m0 = blockIdx.y * 64;
    const int n0 = blockIdx.x * 64;

    const int mt     = wave & 3;
    const int ntbase = (wave >> 2) * 2;

    v8f acc0 = {}; v8f acc1 = {};

    const int lrow = tid >> 2;        // 0..63
    const int lcol = (tid & 3) * 8;   // 0,8,16,24

    for (int k0 = 0; k0 < K; k0 += 32) {
        *(uint4*)&As[lrow][lcol] = *(const uint4*)(A + (size_t)(m0 + lrow) * K + k0 + lcol);
        *(uint4*)&Ws[lrow][lcol] = *(const uint4*)(W + (size_t)(n0 + lrow) * K + k0 + lcol);
        __syncthreads();

        AFrag a;
#pragma unroll
        for (int i = 0; i < 8; ++i)
            a.u[i] = *(const uint32_t*)&As[mt * 16 + lan16][a_frag_k(i, half)];

        BFrag b0, b1;
#pragma unroll
        for (int i = 0; i < 8; ++i) {
            const int kk = b_frag_k(i, half);
            b0.u[i] = *(const uint32_t*)&Ws[ntbase * 16 + lan16][kk];
            b1.u[i] = *(const uint32_t*)&Ws[(ntbase + 1) * 16 + lan16][kk];
        }
        acc0 = __builtin_amdgcn_wmma_f32_16x16x32_bf16(false, a.v, false, b0.v,
                                                       (short)0, acc0, false, false);
        acc1 = __builtin_amdgcn_wmma_f32_16x16x32_bf16(false, a.v, false, b1.v,
                                                       (short)0, acc1, false, false);
        __syncthreads();
    }

#pragma unroll
    for (int r = 0; r < 8; ++r) {
        const int row = m0 + mt * 16 + r + 8 * half;
        const int c0  = n0 + ntbase * 16 + lan16;
        store_out(&C[(size_t)row * N + c0],      (acc0[r] + bias[c0])      * scale);
        store_out(&C[(size_t)row * N + c0 + 16], (acc1[r] + bias[c0 + 16]) * scale);
    }
}

// -------------------------------------------------------- flash attention (bf16)
// One block owns one (b,h) and 8 query tiles (one per wave); 64-key chunks.
// K tile staged by the Tensor Data Mover; V tile staged transposed by VALU.
// Q was pre-scaled by 1/sqrt(DH) in the projection GEMM.
#define SKT  64
#define KPAD 2   // Ks row stride 66 u16 (TDM pad: +1 dword per 32 dwords)
#define VPAD 2   // Vt row stride 66 u16

__global__ void __launch_bounds__(256)
attn_kernel(const uint16_t* __restrict__ Q, const uint16_t* __restrict__ Kb,
            const uint16_t* __restrict__ Vb, uint16_t* __restrict__ O) {
    __shared__ __align__(16) uint16_t Ks[SKT][64 + KPAD];    // [sk][dh]
    __shared__ __align__(16) uint16_t Vt[64][SKT + VPAD];    // [dh][sk]
    __shared__ __align__(16) uint16_t Pl[8][16 * SKT];       // per-wave P staging

    const int tid   = threadIdx.x;
    const int lane  = tid & 31;
    const int wave  = tid >> 5;
    const int half  = lane >> 4;
    const int lan16 = lane & 15;

    const int nqg = (S_ / 16) / 8;
    const int qg  = blockIdx.x % nqg;
    const int h   = (blockIdx.x / nqg) % H_;
    const int b   = blockIdx.x / (nqg * H_);
    const int qt  = qg * 8 + wave;

    const uint16_t* qp = Q  + ((size_t)(b * S_ + qt * 16)) * D_ + h * DH_;
    const uint16_t* kp = Kb + ((size_t)b * S_) * D_ + h * DH_;
    const uint16_t* vp = Vb + ((size_t)b * S_) * D_ + h * DH_;

    AFrag qf[2];
#pragma unroll
    for (int c = 0; c < 2; ++c)
#pragma unroll
        for (int i = 0; i < 8; ++i)
            qf[c].u[i] = *(const uint32_t*)(qp + (size_t)lan16 * D_ +
                                            c * 32 + a_frag_k(i, half));

    v8f o0 = {}, o1 = {}, o2 = {}, o3 = {};
    float mrow[8], lsum[8];
#pragma unroll
    for (int r = 0; r < 8; ++r) { mrow[r] = -1e30f; lsum[r] = 0.f; }

    uint16_t* pl = &Pl[wave][0];

    // TDM descriptor: 64x64 bf16 tile, row stride D_, LDS pad +1dw / 32dw
    const uint32_t ks_lds = (uint32_t)(uintptr_t)&Ks[0][0];
    const i32x8 g1 = { (int)0x01110000,        // data_size=2B, pad_en, interval=32dw, amount=1dw
                       (int)(64u << 16),       // tensor_dim0 = 64
                       (int)(64u << 16),       // tensor_dim1 = 64
                       (int)(64u << 16),       // tile_dim0 = 64
                       (int)SKT,               // tile_dim1 = 64, tile_dim2 = 0
                       (int)D_,                // tensor_dim0_stride = 1024
                       0, 0 };
    const i32x4 gz4 = { 0, 0, 0, 0 };
    const i32x8 gz8 = { 0, 0, 0, 0, 0, 0, 0, 0 };

    // V staging split: thread -> (sk row 0..63, 16 dh cols)
    const int svk = tid >> 2;
    const int svd = (tid & 3) * 16;

    for (int sk0 = 0; sk0 < S_; sk0 += SKT) {
        __syncthreads();

        if (wave == 0) {
            const uint64_t ga = (uint64_t)(uintptr_t)(kp + (size_t)sk0 * D_);
            u32x4 g0 = { 1u, ks_lds, (uint32_t)ga,
                         ((uint32_t)(ga >> 32) & 0x01FFFFFFu) | 0x80000000u };  // type=2
            __builtin_amdgcn_tensor_load_to_lds(g0, g1, gz4, gz4, gz8, 0);
        }
        // stage V transposed: Vt[dh][sk]
        {
            union { uint4 q; uint16_t s[8]; } vv;
            const uint16_t* vg = vp + (size_t)(sk0 + svk) * D_ + svd;
            vv.q = *(const uint4*)vg;
#pragma unroll
            for (int j = 0; j < 8; ++j) Vt[svd + j][svk] = vv.s[j];
            vv.q = *(const uint4*)(vg + 8);
#pragma unroll
            for (int j = 0; j < 8; ++j) Vt[svd + 8 + j][svk] = vv.s[j];
        }
        if (wave == 0) __builtin_amdgcn_s_wait_tensorcnt(0);
        __syncthreads();

        // ---- logits: four 16x16 C tiles (keys sk0 + t*16)
        v8f s[4] = { {}, {}, {}, {} };
#pragma unroll
        for (int c = 0; c < 2; ++c) {
#pragma unroll
            for (int t = 0; t < 4; ++t) {
                BFrag kf;
#pragma unroll
                for (int i = 0; i < 8; ++i)
                    kf.u[i] = *(const uint32_t*)&Ks[t * 16 + lan16][c * 32 + b_frag_k(i, half)];
                s[t] = __builtin_amdgcn_wmma_f32_16x16x32_bf16(false, qf[c].v, false, kf.v,
                                                               (short)0, s[t], false, false);
            }
        }

        // ---- online softmax: one cross-lane max round per 64 keys
        float alpha[8];
#pragma unroll
        for (int r = 0; r < 8; ++r) {
            const float mx = red_max16(fmaxf(fmaxf(s[0][r], s[1][r]),
                                             fmaxf(s[2][r], s[3][r])));
            const float mnew = fmaxf(mrow[r], mx);
            alpha[r] = __expf(mrow[r] - mnew);
            float e0 = __expf(s[0][r] - mnew), e1 = __expf(s[1][r] - mnew);
            float e2 = __expf(s[2][r] - mnew), e3 = __expf(s[3][r] - mnew);
            lsum[r] = lsum[r] * alpha[r] + ((e0 + e1) + (e2 + e3));
            mrow[r] = mnew;
            const int prow = (r + 8 * half) * SKT + lan16;
            pl[prow]      = f2bf(e0);
            pl[prow + 16] = f2bf(e1);
            pl[prow + 32] = f2bf(e2);
            pl[prow + 48] = f2bf(e3);
        }
#pragma unroll
        for (int r = 0; r < 8; ++r) {
            o0[r] *= alpha[r]; o1[r] *= alpha[r];
            o2[r] *= alpha[r]; o3[r] *= alpha[r];
        }
        asm volatile("s_wait_dscnt 0" ::: "memory");  // same-wave cross-lane LDS RAW

        AFrag pf0, pf1;
#pragma unroll
        for (int i = 0; i < 8; ++i) {
            const int kk = a_frag_k(i, half);
            pf0.u[i] = *(const uint32_t*)&pl[lan16 * SKT + kk];
            pf1.u[i] = *(const uint32_t*)&pl[lan16 * SKT + 32 + kk];
        }
        asm volatile("" ::: "memory");

        // ---- O += P @ V : 4 dh-chunks x 2 key-chunks
        v8f* oacc[4] = { &o0, &o1, &o2, &o3 };
#pragma unroll
        for (int c = 0; c < 4; ++c) {
            BFrag vf0, vf1;
#pragma unroll
            for (int i = 0; i < 8; ++i) {
                const int kk = b_frag_k(i, half);
                vf0.u[i] = *(const uint32_t*)&Vt[c * 16 + lan16][kk];
                vf1.u[i] = *(const uint32_t*)&Vt[c * 16 + lan16][32 + kk];
            }
            *oacc[c] = __builtin_amdgcn_wmma_f32_16x16x32_bf16(false, pf0.v, false, vf0.v,
                                                               (short)0, *oacc[c], false, false);
            *oacc[c] = __builtin_amdgcn_wmma_f32_16x16x32_bf16(false, pf1.v, false, vf1.v,
                                                               (short)0, *oacc[c], false, false);
        }
    }

    // ---- final: single cross-lane sum reduction, normalize, store bf16
#pragma unroll
    for (int r = 0; r < 8; ++r) lsum[r] = red_sum16(lsum[r]);
    float inv[8];
#pragma unroll
    for (int r = 0; r < 8; ++r) inv[r] = 1.f / lsum[r];
#pragma unroll
    for (int r = 0; r < 8; ++r) {
        const size_t row = (size_t)(b * S_ + qt * 16 + r + 8 * half);
        uint16_t* op = O + row * D_ + h * DH_ + lan16;
        op[0]  = f2bf(o0[r] * inv[r]);
        op[16] = f2bf(o1[r] * inv[r]);
        op[32] = f2bf(o2[r] * inv[r]);
        op[48] = f2bf(o3[r] * inv[r]);
    }
}

// --------------------------------------------------------------------- launch
extern "C" void kernel_launch(void* const* d_in, const int* in_sizes, int n_in,
                              void* d_out, int out_size, void* d_ws, size_t ws_size,
                              hipStream_t stream) {
    (void)in_sizes; (void)n_in; (void)out_size; (void)ws_size;
    const float* x  = (const float*)d_in[0];
    const float* Wq = (const float*)d_in[1];
    const float* bq = (const float*)d_in[2];
    const float* Wk = (const float*)d_in[3];
    const float* bk = (const float*)d_in[4];
    const float* Wv = (const float*)d_in[5];
    const float* bv = (const float*)d_in[6];
    const float* Wo = (const float*)d_in[7];
    const float* bo = (const float*)d_in[8];
    float* out = (float*)d_out;

    const size_t M = (size_t)B_ * S_;          // 8192
    uint8_t* ws = (uint8_t*)d_ws;
    uint16_t* xb  = (uint16_t*)ws;  ws += M * D_ * 2;
    uint16_t* Wqb = (uint16_t*)ws;  ws += (size_t)D_ * D_ * 2;
    uint16_t* Wkb = (uint16_t*)ws;  ws += (size_t)D_ * D_ * 2;
    uint16_t* Wvb = (uint16_t*)ws;  ws += (size_t)D_ * D_ * 2;
    uint16_t* Wob = (uint16_t*)ws;  ws += (size_t)D_ * D_ * 2;
    uint16_t* qb  = (uint16_t*)ws;  ws += M * D_ * 2;
    uint16_t* kb  = (uint16_t*)ws;  ws += M * D_ * 2;
    uint16_t* vb  = (uint16_t*)ws;  ws += M * D_ * 2;
    uint16_t* ab  = (uint16_t*)ws;  ws += M * D_ * 2;

    cvt_bf16_kernel<<<2048, 256, 0, stream>>>(x,  xb,  (int)(M * D_));
    cvt_bf16_kernel<<<1024, 256, 0, stream>>>(Wq, Wqb, D_ * D_);
    cvt_bf16_kernel<<<1024, 256, 0, stream>>>(Wk, Wkb, D_ * D_);
    cvt_bf16_kernel<<<1024, 256, 0, stream>>>(Wv, Wvb, D_ * D_);
    cvt_bf16_kernel<<<1024, 256, 0, stream>>>(Wo, Wob, D_ * D_);

    dim3 ggrid(D_ / 64, (unsigned)(M / 64));
    // Q pre-scaled by 1/sqrt(DH) = 0.125 (exact power of two)
    gemm_xWT_bias<uint16_t><<<ggrid, 256, 0, stream>>>(xb, Wqb, bq, qb, (int)M, D_, D_, 0.125f);
    gemm_xWT_bias<uint16_t><<<ggrid, 256, 0, stream>>>(xb, Wkb, bk, kb, (int)M, D_, D_, 1.0f);
    gemm_xWT_bias<uint16_t><<<ggrid, 256, 0, stream>>>(xb, Wvb, bv, vb, (int)M, D_, D_, 1.0f);

    attn_kernel<<<B_ * H_ * ((S_ / 16) / 8), 256, 0, stream>>>(qb, kb, vb, ab);

    gemm_xWT_bias<float><<<ggrid, 256, 0, stream>>>(ab, Wob, bo, out, (int)M, D_, D_, 1.0f);
}
